// MixtureOfExpertsAdapter_20761871909269
// MI455X (gfx1250) — matrix operand order repaired and checked by reading, not compiled, not run on hardware
//
#include <hip/hip_runtime.h>
#include <math.h>

// ---------------- problem constants (match reference shapes) ----------------
constexpr int D = 2048;   // hidden dim
constexpr int E = 8;      // experts
constexpr int R = 64;     // LoRA rank
constexpr int O = 2048;   // out dim
constexpr float SCALING = 1.0f / 64.0f;   // 1/RANK

#define DEV __device__ __forceinline__

typedef __attribute__((ext_vector_type(16))) __bf16 v16bf;
typedef __attribute__((ext_vector_type(8)))  float  v8f;

DEV float4 ldg4(const float* p) { return *reinterpret_cast<const float4*>(p); }

DEV v16bf pack16(float4 a, float4 b, float4 c, float4 d) {
  v16bf r;
  r[0]  = (__bf16)a.x; r[1]  = (__bf16)a.y; r[2]  = (__bf16)a.z; r[3]  = (__bf16)a.w;
  r[4]  = (__bf16)b.x; r[5]  = (__bf16)b.y; r[6]  = (__bf16)b.z; r[7]  = (__bf16)b.w;
  r[8]  = (__bf16)c.x; r[9]  = (__bf16)c.y; r[10] = (__bf16)c.z; r[11] = (__bf16)c.w;
  r[12] = (__bf16)d.x; r[13] = (__bf16)d.y; r[14] = (__bf16)d.z; r[15] = (__bf16)d.w;
  return r;
}

// ---------------- kernel 1: zero the atomic counters ----------------
__global__ void moe_init(int* counts, int* cursors) {
  if (threadIdx.x < E) { counts[threadIdx.x] = 0; cursors[threadIdx.x] = 0; }
}

// ---------------- kernel 2: top-1 router (one wave32 per token) ----------------
__global__ __launch_bounds__(256) void moe_router(
    const float* __restrict__ x, const float* __restrict__ Wr,
    const float* __restrict__ br, float* __restrict__ wt,
    int* __restrict__ idx, int* __restrict__ counts, int N) {
  int w = threadIdx.x >> 5, lane = threadIdx.x & 31;
  int tok = blockIdx.x * 8 + w;
  if (tok >= N) return;
  const float* xr = x + (size_t)tok * D;
  float acc[E];
#pragma unroll
  for (int e = 0; e < E; ++e) acc[e] = 0.f;
  for (int k = 0; k < D / 32; ++k) {         // lane-strided, coalesced
    int d = k * 32 + lane;
    float xv = xr[d];
#pragma unroll
    for (int e = 0; e < E; ++e) acc[e] = fmaf(xv, Wr[e * D + d], acc[e]);
  }
#pragma unroll
  for (int e = 0; e < E; ++e) {
#pragma unroll
    for (int off = 16; off > 0; off >>= 1) acc[e] += __shfl_xor(acc[e], off, 32);
  }
  if (lane == 0) {
    float logit[E]; float mx = -3.4e38f; int best = 0;
#pragma unroll
    for (int e = 0; e < E; ++e) {
      logit[e] = acc[e] + br[e];
      if (logit[e] > mx) { mx = logit[e]; best = e; }   // first-max tie-break
    }
    float s = 0.f;
#pragma unroll
    for (int e = 0; e < E; ++e) s += expf(logit[e] - mx);
    wt[tok]  = 1.0f / s;     // softmax prob of the winning expert
    idx[tok] = best;
    atomicAdd(&counts[best], 1);
  }
}

// ---------------- kernel 3: exclusive prefix over 8 counts ----------------
__global__ void moe_offsets(const int* __restrict__ counts, int* __restrict__ offs) {
  if (threadIdx.x == 0) {
    int a = 0; offs[0] = 0;
    for (int e = 0; e < E; ++e) { a += counts[e]; offs[e + 1] = a; }
  }
}

// ---------------- kernel 4: bucket tokens by expert ----------------
__global__ void moe_scatter(const int* __restrict__ idx, const int* __restrict__ offs,
                            int* cursors, int* __restrict__ tlist, int N) {
  int t = blockIdx.x * blockDim.x + threadIdx.x;
  if (t >= N) return;
  int e = idx[t];
  int p = atomicAdd(&cursors[e], 1);
  tlist[offs[e] + p] = t;
}

// ---------------- kernel 5: grouped LoRA GEMM, one block per 16-token tile ----
// stage 1: h[16 x 64] = Xtile[16 x 2048] @ A_e^T        (bf16 WMMA, f32 acc)
// stage 2: y[16 x 2048] = h @ B_e^T, * SCALING*weight    (bf16 WMMA, f32 acc)
__global__ __launch_bounds__(256) void moe_lora_wmma(
    const float* __restrict__ x, const float* __restrict__ A,
    const float* __restrict__ Bw, const float* __restrict__ wt,
    const int* __restrict__ tlist, const int* __restrict__ offs,
    float* __restrict__ out) {
  __shared__ int   s_tok[16];
  __shared__ float s_wt[16];
  __shared__ float h_lds[2][16][R + 4];   // [k-half][token row][r], padded vs bank conflicts

  // map flat tile id -> (expert e, tile t within expert); uniform across block
  int flat = blockIdx.x;
  int e = -1, t = 0, accT = 0, ne = 0;
#pragma unroll
  for (int i = 0; i < E; ++i) {
    int cnt = offs[i + 1] - offs[i];
    int te = (cnt + 15) >> 4;
    if (e < 0 && flat < accT + te) { e = i; t = flat - accT; ne = cnt; }
    accT += te;
  }
  if (e < 0) return;   // uniform early-exit: EXEC stays all-ones for WMMA blocks

  if (threadIdx.x < 16) {
    int m = threadIdx.x;
    bool valid = (t * 16 + m) < ne;
    int g = offs[e] + t * 16 + m;
    int tok = tlist[valid ? g : offs[e]];   // clamp pad slots to a real token
    s_tok[m] = tok;
    s_wt[m]  = valid ? wt[tok] : 0.0f;
  }
  __syncthreads();

  int w = threadIdx.x >> 5, lane = threadIdx.x & 31;
  int m = lane & 15, hi = lane >> 4;

  // ---- stage 1: waves split as (r-tile = w&3) x (K-half = w>>2) ----
  {
    int rt = w & 3, kh = w >> 2;
    int rcol = rt * 16 + m;                              // B-frag column n -> r index
    const float* arow = A + ((size_t)e * R + rcol) * D;  // A[e, r, :]
    const float* xrow = x + (size_t)s_tok[m] * D;        // gathered token row (A-frag row m)
    v8f acc = {};
    for (int kk = 0; kk < (D / 2) / 32; ++kk) {
      int kbase = kh * (D / 2) + kk * 32;
      // A-frag (16-bit 16x32): lane m, elem i -> K = (i>>3)*16 + hi*8 + (i&7)
      int da = kbase + hi * 8;
      v16bf af = pack16(ldg4(xrow + da),      ldg4(xrow + da + 4),
                        ldg4(xrow + da + 16), ldg4(xrow + da + 20));
      // B-frag (16-bit 32x16): lane col n, elem i -> K = (i>>3)*16 + hi*8 + (i&7)
      int db = kbase + hi * 8;
      v16bf bf = pack16(ldg4(arow + db),      ldg4(arow + db + 4),
                        ldg4(arow + db + 16), ldg4(arow + db + 20));
      acc = __builtin_amdgcn_wmma_f32_16x16x32_bf16(false, af, false, bf,
                                                    (short)0, acc, false, false);
    }
    // C/D layout: lane col n = m_lane&15, VGPR v -> row v + 8*hi
#pragma unroll
    for (int v = 0; v < 8; ++v) h_lds[kh][v + 8 * hi][rcol] = acc[v];
  }
  __syncthreads();

  // ---- stage 2: build h A-frags once (fold K-half reduction into the read) ----
  v16bf haf0, haf1;
#pragma unroll
  for (int i = 0; i < 16; ++i) {
    int k0 = ((i >> 3) * 16) + hi * 8 + (i & 7);
    haf0[i] = (__bf16)(h_lds[0][m][k0]      + h_lds[1][m][k0]);
    haf1[i] = (__bf16)(h_lds[0][m][k0 + 32] + h_lds[1][m][k0 + 32]);
  }
  for (int ot = w; ot < O / 16; ot += 8) {
    int ocol = ot * 16 + m;
    const float* brow = Bw + ((size_t)e * O + ocol) * R;   // B[e, o, :], 64 floats
    int r0 = hi * 8;
    v16bf b0 = pack16(ldg4(brow + r0),          ldg4(brow + r0 + 4),
                      ldg4(brow + r0 + 16),     ldg4(brow + r0 + 20));
    v16bf b1 = pack16(ldg4(brow + 32 + r0),     ldg4(brow + 32 + r0 + 4),
                      ldg4(brow + 32 + r0 + 16), ldg4(brow + 32 + r0 + 20));
    v8f acc = {};
    acc = __builtin_amdgcn_wmma_f32_16x16x32_bf16(false, haf0, false, b0,
                                                  (short)0, acc, false, false);
    acc = __builtin_amdgcn_wmma_f32_16x16x32_bf16(false, haf1, false, b1,
                                                  (short)0, acc, false, false);
#pragma unroll
    for (int v = 0; v < 8; ++v) {
      int mr = v + 8 * hi;
      if ((t * 16 + mr) < ne)   // mask pad slots: never touch other tokens' rows
        out[(size_t)s_tok[mr] * O + ocol] = acc[v] * (SCALING * s_wt[mr]);
    }
  }
}

// ---------------- host launcher ----------------
extern "C" void kernel_launch(void* const* d_in, const int* in_sizes, int n_in,
                              void* d_out, int out_size, void* d_ws, size_t ws_size,
                              hipStream_t stream) {
  (void)n_in; (void)out_size; (void)ws_size;
  const float* x  = (const float*)d_in[0];   // [B,S,D]
  const float* A  = (const float*)d_in[1];   // [E,R,D]
  const float* Bw = (const float*)d_in[2];   // [E,O,R]
  const float* Wr = (const float*)d_in[3];   // [E,D]
  const float* br = (const float*)d_in[4];   // [E]
  float* out = (float*)d_out;

  int N = in_sizes[0] / D;                   // 8192 tokens

  // workspace layout (all 4-byte types), ~100 KB total
  float* wt     = (float*)d_ws;              // [N]
  int*   idx    = (int*)d_ws + N;            // [N]
  int*   tlist  = idx + N;                   // [N]
  int*   counts = tlist + N;                 // [E]
  int*   offsp  = counts + E;                // [E+1]
  int*   cursors= offsp + (E + 1);           // [E]

  moe_init   <<<1, 64, 0, stream>>>(counts, cursors);
  moe_router <<<(N + 7) / 8, 256, 0, stream>>>(x, Wr, br, wt, idx, counts, N);
  moe_offsets<<<1, 1, 0, stream>>>(counts, offsp);
  moe_scatter<<<(N + 255) / 256, 256, 0, stream>>>(idx, offsp, cursors, tlist, N);

  int maxTiles = N / 16 + E;                 // upper bound on sum of ceil(n_e/16)
  moe_lora_wmma<<<maxTiles, 256, 0, stream>>>(x, A, Bw, wt, tlist, offsp, out);
}
